// GeneTreeGIN_3281355014346
// MI455X (gfx1250) — matrix operand order repaired and checked by reading, not compiled, not run on hardware
//
#include <hip/hip_runtime.h>

typedef __attribute__((ext_vector_type(16))) __bf16 v16bf;
typedef __attribute__((ext_vector_type(8)))  __bf16 v8bf;
typedef __attribute__((ext_vector_type(4)))  __bf16 v4bf;
typedef __attribute__((ext_vector_type(8)))  float  v8f;

#define GG   800
#define SS   100
#define NPT  199
#define NN   (GG * NPT)      // 159200
#define DD   128
#define ESP  (2 * SS - 2)    // 198
#define GS   (GG * SS)       // 80000
#define LEAF 99
#define NPAD 159232          // 2488 * 64  (row padding so GEMM tiles need no guards)

// ---------------- weight transpose + bf16 convert: wt[n*128+k] = w[k*128+n] ----
__global__ void k_wt(const float* __restrict__ w, __bf16* __restrict__ wt) {
    int idx = blockIdx.x * 256 + threadIdx.x;       // 16384 threads
    int n = idx >> 7, k = idx & 127;
    wt[n * 128 + k] = (__bf16)w[k * 128 + n];
}

// ---------------- x = species_emb[emb_ids] -----------------------------------
__global__ void k_gather(const float* __restrict__ emb, const int* __restrict__ ids,
                         float* __restrict__ X) {
    int idx = blockIdx.x * 256 + threadIdx.x;       // NN*32 threads
    int node = idx >> 5, c = (idx & 31) * 4;
    *(float4*)(X + (size_t)node * DD + c) =
        *(const float4*)(emb + (size_t)ids[node] * DD + c);
}

// ---- Hbf16 = bf16((1+eps)*x + parent + children) (heap structure) ------------
__global__ void k_agg(const float* __restrict__ X, __bf16* __restrict__ H,
                      const float* __restrict__ epsp) {
    int idx = blockIdx.x * 256 + threadIdx.x;       // NN*32 threads
    int node = idx >> 5, c = (idx & 31) * 4;
    int g = node / NPT;
    int l = node - g * NPT;
    int base = g * NPT;
    float4 xv = *(const float4*)(X + (size_t)node * DD + c);
    float ax = 0.f, ay = 0.f, az = 0.f, aw = 0.f;
    if (l > 0) {
        const float4 p = *(const float4*)(X + (size_t)(base + ((l - 1) >> 1)) * DD + c);
        ax += p.x; ay += p.y; az += p.z; aw += p.w;
    }
    if (l < LEAF) {
        const float4 c1 = *(const float4*)(X + (size_t)(base + 2 * l + 1) * DD + c);
        const float4 c2 = *(const float4*)(X + (size_t)(base + 2 * l + 2) * DD + c);
        ax += c1.x + c2.x; ay += c1.y + c2.y; az += c1.z + c2.z; aw += c1.w + c2.w;
    }
    float e1 = 1.0f + epsp[0];
    v4bf o;
    o[0] = (__bf16)(e1 * xv.x + ax);
    o[1] = (__bf16)(e1 * xv.y + ay);
    o[2] = (__bf16)(e1 * xv.z + az);
    o[3] = (__bf16)(e1 * xv.w + aw);
    *(v4bf*)(H + (size_t)node * DD + c) = o;
}

// ---------------- T = bf16(relu(H @ W1 + b1)) via WMMA ------------------------
// 256 thr = 8 waves; 64 rows/block; wave w owns 16-col tile w. No guards (NPAD).
__global__ void __launch_bounds__(256) k_gemm_relu(
    const __bf16* __restrict__ H, const __bf16* __restrict__ Wt,
    const float* __restrict__ bias, __bf16* __restrict__ T) {
    __shared__ __bf16 sh[64 * 128];
    const int tid = threadIdx.x;
    const int lane = tid & 31;
    const int wave = tid >> 5;
    const int half = lane >> 4;
    const int l15 = lane & 15;
    const size_t rowBase = (size_t)blockIdx.x * 64;
    const int ncol = wave * 16 + l15;

    v16bf bfrag[4];
#pragma unroll
    for (int ks = 0; ks < 4; ++ks)
        bfrag[ks] = *(const v16bf*)(Wt + (size_t)ncol * 128 + ks * 32 + half * 16);
    const float bval = bias[ncol];

#pragma unroll
    for (int rt = 0; rt < 4; ++rt) {
        const __bf16* arow = H + (rowBase + rt * 16 + l15) * 128;
        v8f acc = {};
#pragma unroll
        for (int ks = 0; ks < 4; ++ks) {
            // A 16x32 bf16: elem i -> k = ks*32 + half*16 + i
            v8bf b0 = *(const v8bf*)(arow + ks * 32 + half * 16);
            v8bf b1 = *(const v8bf*)(arow + ks * 32 + half * 16 + 8);
            v16bf af;
#pragma unroll
            for (int i = 0; i < 8; ++i) { af[i] = b0[i]; af[8 + i] = b1[i]; }
            acc = __builtin_amdgcn_wmma_f32_16x16x32_bf16(
                false, af, false, bfrag[ks], (short)0, acc, false, false);
        }
#pragma unroll
        for (int r = 0; r < 8; ++r) {          // D: m = r + 8*half
            float v = acc[r] + bval;
            sh[(rt * 16 + half * 8 + r) * 128 + ncol] = (__bf16)(v > 0.f ? v : 0.f);
        }
    }
    __syncthreads();
    // coalesced 16B stores of the whole 64x128 bf16 tile
    const v8bf* s = (const v8bf*)sh;
    v8bf* dst = (v8bf*)(T + rowBase * 128);
#pragma unroll
    for (int q = 0; q < 4; ++q) dst[tid * 4 + q] = s[tid * 4 + q];
}

// ---------------- X = LayerNorm(X + T @ W2 + b2) via WMMA + LDS ---------------
__global__ void __launch_bounds__(256) k_gemm_res_ln(
    const __bf16* __restrict__ T, const __bf16* __restrict__ Wt,
    const float* __restrict__ bias, const float* __restrict__ lng,
    const float* __restrict__ lnb, float* __restrict__ X) {
    __shared__ float hs[64 * 128];
    const int tid = threadIdx.x;
    const int lane = tid & 31;
    const int wave = tid >> 5;
    const int half = lane >> 4;
    const int l15 = lane & 15;
    const size_t rowBase = (size_t)blockIdx.x * 64;
    const int ncol = wave * 16 + l15;

    v16bf bfrag[4];
#pragma unroll
    for (int ks = 0; ks < 4; ++ks)
        bfrag[ks] = *(const v16bf*)(Wt + (size_t)ncol * 128 + ks * 32 + half * 16);
    const float bval = bias[ncol];

#pragma unroll
    for (int rt = 0; rt < 4; ++rt) {
        const __bf16* arow = T + (rowBase + rt * 16 + l15) * 128;
        v8f acc = {};
#pragma unroll
        for (int ks = 0; ks < 4; ++ks) {
            v8bf b0 = *(const v8bf*)(arow + ks * 32 + half * 16);
            v8bf b1 = *(const v8bf*)(arow + ks * 32 + half * 16 + 8);
            v16bf af;
#pragma unroll
            for (int i = 0; i < 8; ++i) { af[i] = b0[i]; af[8 + i] = b1[i]; }
            acc = __builtin_amdgcn_wmma_f32_16x16x32_bf16(
                false, af, false, bfrag[ks], (short)0, acc, false, false);
        }
#pragma unroll
        for (int r = 0; r < 8; ++r)
            hs[(rt * 16 + half * 8 + r) * 128 + ncol] = acc[r] + bval;
    }
    __syncthreads();

    // residual + LayerNorm: 4 threads per row, 32 elems each, shuffle-reduce
    const int row = tid >> 2;
    const int part = tid & 3;
    float* xr = X + (rowBase + row) * 128 + part * 32;
    const float* rp = hs + row * 128 + part * 32;
    float4 v[8];
    float s = 0.f;
#pragma unroll
    for (int i = 0; i < 8; ++i) {
        float4 xv = ((const float4*)xr)[i];
        float4 hv = ((const float4*)rp)[i];
        hv.x += xv.x; hv.y += xv.y; hv.z += xv.z; hv.w += xv.w;
        v[i] = hv;
        s += hv.x + hv.y + hv.z + hv.w;
    }
    s += __shfl_xor(s, 1, 32);
    s += __shfl_xor(s, 2, 32);
    const float mean = s * (1.f / 128.f);
    float vs = 0.f;
#pragma unroll
    for (int i = 0; i < 8; ++i) {
        float dx = v[i].x - mean, dy = v[i].y - mean, dz = v[i].z - mean, dw = v[i].w - mean;
        vs += dx * dx + dy * dy + dz * dz + dw * dw;
    }
    vs += __shfl_xor(vs, 1, 32);
    vs += __shfl_xor(vs, 2, 32);
    const float rstd = rsqrtf(vs * (1.f / 128.f) + 1e-5f);
#pragma unroll
    for (int i = 0; i < 8; ++i) {
        float4 gv = ((const float4*)(lng + part * 32))[i];
        float4 bv = ((const float4*)(lnb + part * 32))[i];
        float4 o;
        o.x = (v[i].x - mean) * rstd * gv.x + bv.x;
        o.y = (v[i].y - mean) * rstd * gv.y + bv.y;
        o.z = (v[i].z - mean) * rstd * gv.z + bv.z;
        o.w = (v[i].w - mean) * rstd * gv.w + bv.w;
        ((float4*)xr)[i] = o;
    }
}

// ---------------- per-(gene,species) leaf mean + count (deterministic) -------
__global__ void k_pool_mean(const float* __restrict__ X, const int* __restrict__ seg,
                            float* __restrict__ SP, float* __restrict__ CNT) {
    int idx = blockIdx.x * 256 + threadIdx.x;        // GS*128 threads
    int gs = idx >> 7, c = idx & 127;
    int g = gs / SS;
    float acc = 0.f;
    int cnt = 0;
    int nb = g * NPT;
    for (int l = LEAF; l < NPT; ++l) {
        int node = nb + l;
        if (seg[node] == gs) { acc += X[(size_t)node * DD + c]; ++cnt; }
    }
    SP[(size_t)gs * DD + c] = acc / fmaxf((float)cnt, 1.f);
    if (c == 0) CNT[gs] = (float)cnt;
}

// ---------------- n_eg[e][g] = #species present in clade(e) for gene g -------
__global__ void k_neg(const float* __restrict__ CNT, const unsigned char* __restrict__ clade,
                      float* __restrict__ NEG) {
    int idx = blockIdx.x * 256 + threadIdx.x;
    if (idx >= ESP * GG) return;
    int e = idx / GG, g = idx - e * GG;
    const unsigned char* cl = clade + e * SS;
    const float* cnt = CNT + g * SS;
    float n = 0.f;
    for (int s = 0; s < SS; ++s)
        if (cl[s] && cnt[s] > 0.f) n += 1.f;
    NEG[idx] = n;
}

// ---------------- per-edge mean / unbiased std over genes --------------------
__global__ void k_edge(const float* __restrict__ SP, const float* __restrict__ NEG,
                       const unsigned char* __restrict__ clade, float* __restrict__ out) {
    __shared__ int slist[SS];
    __shared__ int scnt;
    int e = blockIdx.x, d = threadIdx.x;   // 198 blocks x 128 threads
    if (d == 0) {
        int c = 0;
        for (int s = 0; s < SS; ++s) if (clade[e * SS + s]) slist[c++] = s;
        scnt = c;
    }
    __syncthreads();
    int ns = scnt;
    float sum = 0.f, sumsq = 0.f;
    int K = 0;
    for (int g = 0; g < GG; ++g) {
        float ne = NEG[e * GG + g];
        if (ne > 0.f) {
            float acc = 0.f;
            for (int j = 0; j < ns; ++j)
                acc += SP[(size_t)(g * SS + slist[j]) * DD + d];
            float cm = acc / ne;
            sum += cm; sumsq += cm * cm; ++K;
        }
    }
    float Kf = (float)K;
    float mean = sum / fmaxf(Kf, 1.f);
    float var = fmaxf((sumsq - sum * sum / fmaxf(Kf, 1.f)) / fmaxf(Kf - 1.f, 1.f), 0.f);
    float sd = (K > 1) ? sqrtf(var) : 0.f;
    out[e * 256 + d] = (K > 0) ? mean : 0.f;
    out[e * 256 + 128 + d] = (K > 0) ? sd : 0.f;
}

extern "C" void kernel_launch(void* const* d_in, const int* in_sizes, int n_in,
                              void* d_out, int out_size, void* d_ws, size_t ws_size,
                              hipStream_t stream) {
    (void)in_sizes; (void)n_in; (void)out_size; (void)ws_size;
    const float* species_emb = (const float*)d_in[0];
    const float* w1[2]  = {(const float*)d_in[1],  (const float*)d_in[8]};
    const float* b1[2]  = {(const float*)d_in[2],  (const float*)d_in[9]};
    const float* w2[2]  = {(const float*)d_in[3],  (const float*)d_in[10]};
    const float* b2[2]  = {(const float*)d_in[4],  (const float*)d_in[11]};
    const float* eps[2] = {(const float*)d_in[5],  (const float*)d_in[12]};
    const float* lng[2] = {(const float*)d_in[6],  (const float*)d_in[13]};
    const float* lnb[2] = {(const float*)d_in[7],  (const float*)d_in[14]};
    const int* emb_ids  = (const int*)d_in[17];
    const int* leaf_seg = (const int*)d_in[18];
    const unsigned char* clade = (const unsigned char*)d_in[19];
    float* out = (float*)d_out;

    // workspace layout (256B aligned)
    char* ws = (char*)d_ws;
    size_t off = 0;
    auto take = [&](size_t bytes) { void* p = ws + off; off += (bytes + 255) & ~(size_t)255; return p; };
    float*  X   = (float*)take((size_t)NPAD * DD * 4);   // fp32 activations (padded)
    char*   Hb  = (char*)take((size_t)NPAD * DD * 2);    // bf16 GIN pre-MLP
    char*   Tb  = (char*)take((size_t)NPAD * DD * 2);    // bf16 hidden
    __bf16* W1t[2], *W2t[2];
    for (int l = 0; l < 2; ++l) {
        W1t[l] = (__bf16*)take(128 * 128 * 2);
        W2t[l] = (__bf16*)take(128 * 128 * 2);
    }
    __bf16* H  = (__bf16*)Hb;
    __bf16* T  = (__bf16*)Tb;
    // post-GIN overlays: SP (40.96 MB fp32) starts at Hb and spills ~192KB into Tb;
    // CNT/NEG live deeper inside Tb, clear of the spill.
    float* SP  = (float*)Hb;
    float* CNT = (float*)(Tb + (512 << 10));
    float* NEG = (float*)(Tb + (1024 << 10));

    const int g32  = (NN * 32) / 256;                 // 19900
    const int gemB = NPAD / 64;                       // 2488

    for (int l = 0; l < 2; ++l) {
        hipLaunchKernelGGL(k_wt, dim3(64), dim3(256), 0, stream, w1[l], W1t[l]);
        hipLaunchKernelGGL(k_wt, dim3(64), dim3(256), 0, stream, w2[l], W2t[l]);
    }
    hipLaunchKernelGGL(k_gather, dim3(g32), dim3(256), 0, stream, species_emb, emb_ids, X);

    for (int l = 0; l < 2; ++l) {
        hipLaunchKernelGGL(k_agg, dim3(g32), dim3(256), 0, stream, X, H, eps[l]);
        hipLaunchKernelGGL(k_gemm_relu, dim3(gemB), dim3(256), 0, stream,
                           H, W1t[l], b1[l], T);
        hipLaunchKernelGGL(k_gemm_res_ln, dim3(gemB), dim3(256), 0, stream,
                           T, W2t[l], b2[l], lng[l], lnb[l], X);
    }

    hipLaunchKernelGGL(k_pool_mean, dim3((GS * DD) / 256), dim3(256), 0, stream,
                       X, leaf_seg, SP, CNT);
    hipLaunchKernelGGL(k_neg, dim3((ESP * GG + 255) / 256), dim3(256), 0, stream,
                       CNT, clade, NEG);
    hipLaunchKernelGGL(k_edge, dim3(ESP), dim3(128), 0, stream, SP, NEG, clade, out);
}
